// TransfromerBlock_58600533786991
// MI455X (gfx1250) — compile-verified, matrix operand order checked by reading, not run on hardware
//
#include <hip/hip_runtime.h>

// ---------------- CDNA5 WMMA types/helpers ----------------
typedef __attribute__((ext_vector_type(16))) __bf16 v16bf;
typedef __attribute__((ext_vector_type(8)))  __bf16 bf8v;   // 16 bytes
typedef __attribute__((ext_vector_type(8)))  float  v8f;
typedef __attribute__((ext_vector_type(4)))  int    v4i;    // 16 bytes

#define DIM    768
#define HEADS  12
#define HD     64
#define NSEQ   1024
#define BATCH  8
#define MTOT   (BATCH*NSEQ)   // 8192
#define HIDDEN 3072

#define AS1 __attribute__((address_space(1)))
#define AS3 __attribute__((address_space(3)))

#if defined(__AMDGCN__) && __has_builtin(__builtin_amdgcn_global_load_async_to_lds_b128) && __has_builtin(__builtin_amdgcn_s_wait_asynccnt)
#define ASYNC_LDS 1
#endif

__device__ inline __bf16 to_bf16(float f){
  unsigned u = __builtin_bit_cast(unsigned, f);
  unsigned r = u + 0x7FFFu + ((u >> 16) & 1u);   // round-to-nearest-even
  unsigned short h = (unsigned short)(r >> 16);
  return __builtin_bit_cast(__bf16, h);
}

// 16-byte global->LDS copy: async DMA (ASYNCcnt) on CDNA5, sync fallback otherwise
__device__ inline void copy16_g2l(const __bf16* g, __bf16* l){
#ifdef ASYNC_LDS
  __builtin_amdgcn_global_load_async_to_lds_b128(
      (AS1 v4i*)g, (AS3 v4i*)l, 0, 0);
#else
  *(bf8v*)l = *(const bf8v*)g;
#endif
}

// Load a 16x32 bf16 A/B fragment (ISA 7.12.2 layout):
// lane<16 : row = lane, K = {0..7, 16..23}; lane>=16 : row = lane-16, K = {8..15, 24..31}
__device__ inline v16bf load_frag(const __bf16* p, int stride, int lane){
  int r  = lane & 15;
  int ko = (lane & 16) ? 8 : 0;
  bf8v lo = *(const bf8v*)(p + (size_t)r * stride + ko);
  bf8v hi = *(const bf8v*)(p + (size_t)r * stride + ko + 16);
  return __builtin_shufflevector(lo, hi, 0,1,2,3,4,5,6,7,8,9,10,11,12,13,14,15);
}

__device__ inline v8f wmma_bf16(v16bf a, v16bf b, v8f c){
  return __builtin_amdgcn_wmma_f32_16x16x32_bf16(false, a, false, b, (short)0, c, false, false);
}

// ---------------- fp32 -> bf16 weight conversion ----------------
__global__ void cvt_kernel(const float* __restrict__ in, __bf16* __restrict__ out, int n){
  int i = blockIdx.x * 256 + threadIdx.x;
  if (i < n) out[i] = to_bf16(in[i]);
}

// ---------------- LayerNorm (row per block) ----------------
__global__ __launch_bounds__(256) void ln_kernel(const float* __restrict__ x,
    const float* __restrict__ g, const float* __restrict__ b, __bf16* __restrict__ out)
{
  int row = blockIdx.x;
  int tid = threadIdx.x;
  const float* xr = x + (size_t)row * DIM;
  __shared__ float red[256];
  float s = 0.f;
  for (int i = tid; i < DIM; i += 256) s += xr[i];
  red[tid] = s; __syncthreads();
  for (int o = 128; o > 0; o >>= 1){ if (tid < o) red[tid] += red[tid+o]; __syncthreads(); }
  float mu = red[0] / (float)DIM;
  __syncthreads();
  float v = 0.f;
  for (int i = tid; i < DIM; i += 256){ float d = xr[i] - mu; v += d*d; }
  red[tid] = v; __syncthreads();
  for (int o = 128; o > 0; o >>= 1){ if (tid < o) red[tid] += red[tid+o]; __syncthreads(); }
  float rs = rsqrtf(red[0] / (float)DIM + 1e-5f);
  for (int i = tid; i < DIM; i += 256)
    out[(size_t)row * DIM + i] = to_bf16((xr[i] - mu) * rs * g[i] + b[i]);
}

// ---------------- Tiled bf16 WMMA GEMM: D = A[M,K] * B[N,K]^T + epilogue ----------------
// Block tile 128(M) x 256(N) x 32(K), 8 waves in 2x4 grid, each wave 64x64
// (16 WMMA per wave per k-step, 1.0 ds_b128 fragment load per WMMA).
// Double-buffered LDS fed by GLOBAL_LOAD_ASYNC_TO_LDS_B128 (ASYNCcnt-tracked).
constexpr int EPI_QKV  = 0;   // scatter into q (scaled), k, vT buffers (bf16)
constexpr int EPI_GELU = 1;   // +bias, exact GELU, bf16 out
constexpr int EPI_RES  = 2;   // +bias, *ls, +residual, f32 out

template<int EPI>
__global__ __launch_bounds__(256) void gemm_bf16_kernel(
    const __bf16* __restrict__ A, const __bf16* __restrict__ B,
    int M, int N, int K,
    const float* __restrict__ bias, const float* __restrict__ ls,
    const float* __restrict__ res, float* __restrict__ outF,
    __bf16* __restrict__ outB,
    __bf16* __restrict__ qb, __bf16* __restrict__ kb, __bf16* __restrict__ vtb)
{
  __shared__ __bf16 sA[2][128 * 32];   // 16 KB
  __shared__ __bf16 sB[2][256 * 32];   // 32 KB
  int tid  = threadIdx.x;
  int lane = tid & 31;
  int wid  = tid >> 5;
  int wm   = wid & 1;          // 2 waves along M (64 rows each)
  int wn   = wid >> 1;         // 4 waves along N (64 cols each)
  int m0   = blockIdx.y * 128;
  int n0   = blockIdx.x * 256;

  // stage one 128x32 A tile + 256x32 B tile into LDS buffer `buf`
  // 6 x 16B transfers per thread per call
  auto stage = [&](int k0, int buf){
    #pragma unroll
    for (int c = tid; c < 512; c += 256){          // A: 128 rows x 4 chunks
      int row = c >> 2, part = (c & 3) * 8;
      copy16_g2l(&A[(size_t)(m0 + row) * K + k0 + part], &sA[buf][row*32 + part]);
    }
    #pragma unroll
    for (int c = tid; c < 1024; c += 256){         // B: 256 rows x 4 chunks
      int row = c >> 2, part = (c & 3) * 8;
      copy16_g2l(&B[(size_t)(n0 + row) * K + k0 + part], &sB[buf][row*32 + part]);
    }
  };

  v8f acc[4][4] = {};
  const int nk = K >> 5;
  stage(0, 0);
  for (int i = 0; i < nk; ++i){
    int cur = i & 1;
    if (i + 1 < nk){
      stage((i + 1) << 5, cur ^ 1);                // prefetch next tile
#ifdef ASYNC_LDS
      __builtin_amdgcn_s_wait_asynccnt(6);         // 6 in flight for next tile => current done
#endif
    } else {
#ifdef ASYNC_LDS
      __builtin_amdgcn_s_wait_asynccnt(0);
#endif
    }
    __syncthreads();                               // current tile visible to all waves

    v16bf af[4], bfv[4];
    #pragma unroll
    for (int mf = 0; mf < 4; ++mf) af[mf]  = load_frag(&sA[cur][(wm*64 + mf*16)*32], 32, lane);
    #pragma unroll
    for (int nf = 0; nf < 4; ++nf) bfv[nf] = load_frag(&sB[cur][(wn*64 + nf*16)*32], 32, lane);
    #pragma unroll
    for (int mf = 0; mf < 4; ++mf)
      #pragma unroll
      for (int nf = 0; nf < 4; ++nf)
        acc[mf][nf] = wmma_bf16(af[mf], bfv[nf], acc[mf][nf]);
    __syncthreads();                               // done reading cur before it is overwritten
  }

  // epilogue: C frag layout — lane holds col (lane&15), 8 rows per lane half
  int cn = lane & 15, chi = (lane >> 4) & 1;
  #pragma unroll
  for (int mf = 0; mf < 4; ++mf){
    #pragma unroll
    for (int nf = 0; nf < 4; ++nf){
      int n = n0 + wn*64 + nf*16 + cn;
      int mbase = m0 + wm*64 + mf*16 + chi*8;
      #pragma unroll
      for (int v = 0; v < 8; ++v){
        int m = mbase + v;
        float val = acc[mf][nf][v];
        if (EPI == EPI_QKV){
          int which = n / DIM, rem = n % DIM;
          int head = rem / HD, d = rem % HD;
          int b = m / NSEQ, t = m % NSEQ;
          size_t bh = (size_t)(b * HEADS + head);
          if      (which == 0) qb [(bh*NSEQ + t)*HD + d] = to_bf16(val * 0.125f); // 1/sqrt(64)
          else if (which == 1) kb [(bh*NSEQ + t)*HD + d] = to_bf16(val);
          else                 vtb[(bh*HD  + d)*NSEQ + t] = to_bf16(val);          // V transposed
        } else if (EPI == EPI_GELU){
          float xx = val + bias[n];
          float gg = 0.5f * xx * (1.0f + erff(xx * 0.70710678118654752f));
          outB[(size_t)m * N + n] = to_bf16(gg);
        } else {
          float xx = val + bias[n];
          size_t idx = (size_t)m * N + n;
          outF[idx] = res[idx] + ls[n] * xx;
        }
      }
    }
  }
}

// ---------------- Flash attention: 1 wave per 16-query tile ----------------
__global__ __launch_bounds__(32) void attn_kernel(
    const __bf16* __restrict__ qb, const __bf16* __restrict__ kb,
    const __bf16* __restrict__ vtb, __bf16* __restrict__ ob)
{
  int blk  = blockIdx.x;          // B*H*(N/16) = 6144
  int qt   = blk & 63;
  int bh   = blk >> 6;            // 0..95
  int head = bh % HEADS, b = bh / HEADS;
  int lane = threadIdx.x;
  int cn = lane & 15, chi = (lane >> 4) & 1;

  const __bf16* Q  = qb  + (size_t)bh * NSEQ * HD;
  const __bf16* Kp = kb  + (size_t)bh * NSEQ * HD;
  const __bf16* VT = vtb + (size_t)bh * HD * NSEQ;
  int q0 = qt * 16;

  v16bf qf[2];
  qf[0] = load_frag(Q + (size_t)q0*HD + 0,  HD, lane);
  qf[1] = load_frag(Q + (size_t)q0*HD + 32, HD, lane);

  v8f of[4] = {};                 // O accumulator 16x64
  float mrow = -1e30f, lrow = 0.f;
  __shared__ float  sS[16 * 32];
  __shared__ __bf16 sP[16 * 32];
  __shared__ float  sAlpha[16];
  __shared__ float  sL[16];

  for (int j0 = 0; j0 < NSEQ; j0 += 32){
    // S chunk 16x32 = Q(16x64) * K(32x64)^T
    v8f s0 = {}, s1 = {};
    #pragma unroll
    for (int s = 0; s < 2; ++s){
      v16bf kf0 = load_frag(Kp + (size_t)(j0     )*HD + s*32, HD, lane);
      v16bf kf1 = load_frag(Kp + (size_t)(j0 + 16)*HD + s*32, HD, lane);
      s0 = wmma_bf16(qf[s], kf0, s0);
      s1 = wmma_bf16(qf[s], kf1, s1);
    }
    #pragma unroll
    for (int v = 0; v < 8; ++v){
      int r = v + chi*8;
      sS[r*32 + cn]      = s0[v];
      sS[r*32 + 16 + cn] = s1[v];
    }
    __syncthreads();
    // online softmax: lane r < 16 owns row r
    if (lane < 16){
      float mx = mrow;
      for (int jj = 0; jj < 32; ++jj) mx = fmaxf(mx, sS[lane*32 + jj]);
      float alpha = __expf(mrow - mx);
      float sum = 0.f;
      for (int jj = 0; jj < 32; ++jj){
        float e = __expf(sS[lane*32 + jj] - mx);
        sum += e;
        sP[lane*32 + jj] = to_bf16(e);
      }
      lrow = lrow * alpha + sum;
      mrow = mx;
      sAlpha[lane] = alpha;
    }
    __syncthreads();
    // rescale O accumulators per row, then O += P(16x32) * V(32x64)
    float al[8];
    #pragma unroll
    for (int v = 0; v < 8; ++v) al[v] = sAlpha[v + chi*8];
    #pragma unroll
    for (int nf = 0; nf < 4; ++nf)
      #pragma unroll
      for (int v = 0; v < 8; ++v) of[nf][v] *= al[v];
    v16bf pf = load_frag(sP, 32, lane);
    #pragma unroll
    for (int nf = 0; nf < 4; ++nf){
      v16bf vf = load_frag(VT + (size_t)(nf*16)*NSEQ + j0, NSEQ, lane);
      of[nf] = wmma_bf16(pf, vf, of[nf]);
    }
    __syncthreads();
  }

  if (lane < 16) sL[lane] = lrow;
  __syncthreads();
  float li[8];
  #pragma unroll
  for (int v = 0; v < 8; ++v) li[v] = 1.f / sL[v + chi*8];
  #pragma unroll
  for (int nf = 0; nf < 4; ++nf){
    int d = nf*16 + cn;
    #pragma unroll
    for (int v = 0; v < 8; ++v){
      int t = q0 + v + chi*8;
      ob[((size_t)(b*NSEQ + t))*DIM + head*HD + d] = to_bf16(of[nf][v] * li[v]);
    }
  }
}

// ---------------- host-side orchestration ----------------
extern "C" void kernel_launch(void* const* d_in, const int* in_sizes, int n_in,
                              void* d_out, int out_size, void* d_ws, size_t ws_size,
                              hipStream_t stream)
{
  const float* x      = (const float*)d_in[0];
  const float* ln1_g  = (const float*)d_in[1];
  const float* ln1_b  = (const float*)d_in[2];
  const float* w_qkv  = (const float*)d_in[3];
  const float* w_proj = (const float*)d_in[4];
  const float* b_proj = (const float*)d_in[5];
  const float* ls1_g  = (const float*)d_in[6];
  const float* ln2_g  = (const float*)d_in[7];
  const float* ln2_b  = (const float*)d_in[8];
  const float* w_fc1  = (const float*)d_in[9];
  const float* b_fc1  = (const float*)d_in[10];
  const float* w_fc2  = (const float*)d_in[11];
  const float* b_fc2  = (const float*)d_in[12];
  const float* ls2_g  = (const float*)d_in[13];
  float* out = (float*)d_out;

  char* ws = (char*)d_ws;
  size_t off = 0;
  auto carve = [&](size_t bytes) -> char* {
    char* p = ws + off;
    off += (bytes + 255) & ~(size_t)255;
    return p;
  };
  __bf16* wqkv_bf  = (__bf16*)carve((size_t)3*DIM*DIM      * 2);
  __bf16* wproj_bf = (__bf16*)carve((size_t)DIM*DIM        * 2);
  __bf16* wfc1_bf  = (__bf16*)carve((size_t)HIDDEN*DIM     * 2);
  __bf16* wfc2_bf  = (__bf16*)carve((size_t)DIM*HIDDEN     * 2);
  __bf16* h1_bf    = (__bf16*)carve((size_t)MTOT*DIM       * 2);
  __bf16* q_bf     = (__bf16*)carve((size_t)MTOT*DIM       * 2);
  __bf16* k_bf     = (__bf16*)carve((size_t)MTOT*DIM       * 2);
  __bf16* vt_bf    = (__bf16*)carve((size_t)MTOT*DIM       * 2);
  __bf16* o_bf     = (__bf16*)carve((size_t)MTOT*DIM       * 2);
  float*  x1       = (float*) carve((size_t)MTOT*DIM       * 4);
  __bf16* h2_bf    = (__bf16*)carve((size_t)MTOT*DIM       * 2);
  __bf16* fc1_bf   = (__bf16*)carve((size_t)MTOT*HIDDEN    * 2);
  (void)ws_size; (void)in_sizes; (void)n_in; (void)out_size;

  // 1. weights -> bf16
  int nq = 3*DIM*DIM;
  cvt_kernel<<<(nq + 255)/256, 256, 0, stream>>>(w_qkv, wqkv_bf, nq);
  cvt_kernel<<<(DIM*DIM + 255)/256, 256, 0, stream>>>(w_proj, wproj_bf, DIM*DIM);
  cvt_kernel<<<(HIDDEN*DIM + 255)/256, 256, 0, stream>>>(w_fc1, wfc1_bf, HIDDEN*DIM);
  cvt_kernel<<<(DIM*HIDDEN + 255)/256, 256, 0, stream>>>(w_fc2, wfc2_bf, DIM*HIDDEN);

  // 2. LN1
  ln_kernel<<<MTOT, 256, 0, stream>>>(x, ln1_g, ln1_b, h1_bf);

  // 3. QKV GEMM (epilogue scatters q/k/vT)
  dim3 g_qkv(3*DIM/256, MTOT/128);   // 9 x 64
  gemm_bf16_kernel<EPI_QKV><<<g_qkv, 256, 0, stream>>>(h1_bf, wqkv_bf, MTOT, 3*DIM, DIM,
      nullptr, nullptr, nullptr, nullptr, nullptr, q_bf, k_bf, vt_bf);

  // 4. attention
  attn_kernel<<<BATCH*HEADS*(NSEQ/16), 32, 0, stream>>>(q_bf, k_bf, vt_bf, o_bf);

  // 5. proj + ls1 + residual -> x1 (f32)
  dim3 g_proj(DIM/256, MTOT/128);    // 3 x 64
  gemm_bf16_kernel<EPI_RES><<<g_proj, 256, 0, stream>>>(o_bf, wproj_bf, MTOT, DIM, DIM,
      b_proj, ls1_g, x, x1, nullptr, nullptr, nullptr, nullptr);

  // 6. LN2
  ln_kernel<<<MTOT, 256, 0, stream>>>(x1, ln2_g, ln2_b, h2_bf);

  // 7. fc1 + GELU -> bf16
  dim3 g_fc1(HIDDEN/256, MTOT/128);  // 12 x 64
  gemm_bf16_kernel<EPI_GELU><<<g_fc1, 256, 0, stream>>>(h2_bf, wfc1_bf, MTOT, HIDDEN, DIM,
      b_fc1, nullptr, nullptr, nullptr, fc1_bf, nullptr, nullptr, nullptr);

  // 8. fc2 + ls2 + residual -> out (f32)
  dim3 g_fc2(DIM/256, MTOT/128);     // 3 x 64
  gemm_bf16_kernel<EPI_RES><<<g_fc2, 256, 0, stream>>>(fc1_bf, wfc2_bf, MTOT, DIM, HIDDEN,
      b_fc2, ls2_g, x1, out, nullptr, nullptr, nullptr, nullptr);
}
